// single_head_split_window_attention_48387101557134
// MI455X (gfx1250) — matrix operand order repaired and checked by reading, not compiled, not run on hardware
//
#include <hip/hip_runtime.h>

typedef __attribute__((ext_vector_type(16))) __bf16 v16bf;
typedef __attribute__((ext_vector_type(2)))  __bf16 bf16x2;
typedef __attribute__((ext_vector_type(8)))  float  v8f;
typedef __attribute__((ext_vector_type(4)))  float  f32x4;
typedef __attribute__((ext_vector_type(2)))  float  f32x2;
typedef __attribute__((ext_vector_type(4)))  unsigned int u32x4;
typedef __attribute__((ext_vector_type(8)))  unsigned int u32x8;
typedef __attribute__((ext_vector_type(2)))  unsigned int u32x2;

#define HD        128    // channels
#define LWIN      1024   // tokens per window
#define WIN_SIDE  32     // window side
#define HW        65536  // H*W
#define IMG       256    // H == W
#define KROW      136    // K LDS row stride in bf16 (272 B, 16B aligned, bank-spread)
#define VROW      20     // V LDS row stride in u32 (80 B, 16B aligned, bank-spread)

// f32 pair -> packed bf16 (RNE). Plain vector-convert IR: lowers to
// v_cvt_pk_bf16_f32 on targets with native bf16 support.
__device__ __forceinline__ unsigned int pk_bf16(float a, float b) {
  f32x2 v = {a, b};
  bf16x2 r = __builtin_convertvector(v, bf16x2);
  return __builtin_bit_cast(unsigned int, r);
}

__device__ __forceinline__ int tok_index(int wy, int wx, int t, int shift) {
  int ly = t >> 5, lx = t & 31;
  int row = (wy * WIN_SIDE + ly + shift) & (IMG - 1);
  int col = (wx * WIN_SIDE + lx + shift) & (IMG - 1);
  return row * IMG + col;
}

__device__ __forceinline__ v16bf ld_bf16x16(const unsigned short* p0,
                                            const unsigned short* p1) {
  u32x4 lo = *(const u32x4*)p0;
  u32x4 hi = *(const u32x4*)p1;
  u32x8 w  = __builtin_shufflevector(lo, hi, 0, 1, 2, 3, 4, 5, 6, 7);
  return __builtin_bit_cast(v16bf, w);
}

// Stage one 32-key tile of K (bf16 row-major) and V (bf16 transposed, key-pair
// packed) into the given LDS buffers. 256 threads cooperate.
__device__ __forceinline__ void stage_tile(int tid, int wy, int wx, int shift,
                                           size_t bOff, int kbase,
                                           const float* __restrict__ Kg,
                                           const float* __restrict__ Vg,
                                           unsigned short* Kbuf,
                                           unsigned int* Vbuf) {
#pragma unroll
  for (int i = 0; i < 4; ++i) {
    int idx = tid + i * 256;               // 1024 float4 quads
    int key = idx >> 5;
    int ch4 = (idx & 31) << 2;
    int tk  = tok_index(wy, wx, kbase + key, shift);
    f32x4 d = *(const f32x4*)(Kg + (bOff + tk) * HD + ch4);
    u32x2 pw;
    pw[0] = pk_bf16(d[0], d[1]);
    pw[1] = pk_bf16(d[2], d[3]);
    *(u32x2*)&Kbuf[key * KROW + ch4] = pw;
  }
#pragma unroll
  for (int i = 0; i < 8; ++i) {
    int idx = tid + i * 256;               // 2048 pair-words
    int ch = idx & 127;
    int kp = idx >> 7;
    int t0 = tok_index(wy, wx, kbase + 2 * kp + 0, shift);
    int t1 = tok_index(wy, wx, kbase + 2 * kp + 1, shift);
    float f0 = Vg[(bOff + t0) * HD + ch];
    float f1 = Vg[(bOff + t1) * HD + ch];
    Vbuf[ch * VROW + kp] = pk_bf16(f0, f1);
  }
}

__global__ __launch_bounds__(256)
void swin_flash_attn_wmma(const float* __restrict__ Qg,
                          const float* __restrict__ Kg,
                          const float* __restrict__ Vg,
                          const float* __restrict__ Mask,
                          const int*   __restrict__ wsPtr,
                          float*       __restrict__ Out) {
  // Double-buffered tiles: 2 x (8704 + 10240) B ~ 38 KB of 320 KB/WGP
  __shared__ __align__(16) unsigned short Klds[2][32 * KROW];
  __shared__ __align__(16) unsigned int   Vt[2][HD * VROW];

  const int tid  = threadIdx.x;
  const int lane = tid & 31;
  const int wv   = tid >> 5;       // wave 0..7
  const int q16  = lane & 15;      // lane%16: q (as B-N / D-N) or key row (as A-M)
  const int half = lane >> 4;

  const int blk   = blockIdx.x;
  const int win   = blk >> 3;      // 256 windows
  const int qtile = blk & 7;       // 8 query tiles of 128 rows
  const int b     = win >> 6;
  const int kk    = win & 63;
  const int wy    = kk >> 3;
  const int wx    = kk & 7;

  const int ws    = wsPtr[0];
  const int shift = ws ? (WIN_SIDE / 2) : 0;

  const size_t bOff     = (size_t)b * HW;
  const float inv_scale = 0.08838834764831845f;  // 1/sqrt(128)

  // ---- per-lane query row, preload Q as WMMA B operand (4 channel chunks) ----
  const int qlocal = qtile * 128 + wv * 16 + q16;
  const int qtok   = tok_index(wy, wx, qlocal, shift);
  const float* qrow = Qg + (bOff + qtok) * HD;

  v16bf Qb[4];
#pragma unroll
  for (int c = 0; c < 4; ++c) {
    int base = c * 32 + 8 * half;            // B K-dim: 8 channels at 8*half, 8 at 16+8*half
    f32x4 f0 = *(const f32x4*)(qrow + base + 0);
    f32x4 f1 = *(const f32x4*)(qrow + base + 4);
    f32x4 f2 = *(const f32x4*)(qrow + base + 16);
    f32x4 f3 = *(const f32x4*)(qrow + base + 20);
    u32x8 w;
    w[0] = pk_bf16(f0[0], f0[1]); w[1] = pk_bf16(f0[2], f0[3]);
    w[2] = pk_bf16(f1[0], f1[1]); w[3] = pk_bf16(f1[2], f1[3]);
    w[4] = pk_bf16(f2[0], f2[1]); w[5] = pk_bf16(f2[2], f2[3]);
    w[6] = pk_bf16(f3[0], f3[1]); w[7] = pk_bf16(f3[2], f3[3]);
    Qb[c] = __builtin_bit_cast(v16bf, w);
  }

  const float* mrow = Mask + ((size_t)kk * LWIN + qlocal) * LWIN;

  v8f vzero = {0.f, 0.f, 0.f, 0.f, 0.f, 0.f, 0.f, 0.f};
  v8f Oacc[8];
#pragma unroll
  for (int c = 0; c < 8; ++c) Oacc[c] = vzero;
  float m_run = -3.0e38f;
  float l_run = 0.0f;

  // Prologue: stage tile 0 into buffer 0
  stage_tile(tid, wy, wx, shift, bOff, 0, Kg, Vg, Klds[0], Vt[0]);

  for (int kt = 0; kt < 32; ++kt) {
    const int cur   = kt & 1;
    const int kbase = kt * 32;
    // One barrier per iteration:
    //  - makes buffer `cur` (staged last iteration) visible to all waves
    //  - guarantees reads of buffer `cur^1` from iteration kt-1 are complete
    //    before it is overwritten below
    __syncthreads();
    if (kt + 1 < 32) {
      stage_tile(tid, wy, wx, shift, bOff, kbase + 32, Kg, Vg,
                 Klds[cur ^ 1], Vt[cur ^ 1]);
    }

    const unsigned short* Kb = Klds[cur];
    const unsigned int*   Vb = Vt[cur];

    // ---- S' = K x Q^T : D has q in lanes, key rows in VGPRs ----
    v8f s0 = vzero, s1 = vzero;
#pragma unroll
    for (int c = 0; c < 4; ++c) {
      int cb = c * 32;
      v16bf a0 = ld_bf16x16(&Kb[q16 * KROW + cb + 8 * half],
                            &Kb[q16 * KROW + cb + 16 + 8 * half]);
      s0 = __builtin_amdgcn_wmma_f32_16x16x32_bf16(false, a0, false, Qb[c],
                                                   (short)0, s0, false, false);
    }
#pragma unroll
    for (int c = 0; c < 4; ++c) {
      int cb = c * 32;
      v16bf a1 = ld_bf16x16(&Kb[(16 + q16) * KROW + cb + 8 * half],
                            &Kb[(16 + q16) * KROW + cb + 16 + 8 * half]);
      s1 = __builtin_amdgcn_wmma_f32_16x16x32_bf16(false, a1, false, Qb[c],
                                                   (short)0, s1, false, false);
    }

    // ---- scale + mask (acc VGPR r <-> key = sub*16 + 8*half + r) ----
    float sc0[8], sc1[8];
    if (ws) {
      const float* mp0 = mrow + kbase + 8 * half;
      const float* mp1 = mrow + kbase + 16 + 8 * half;
      f32x4 ma = *(const f32x4*)(mp0);
      f32x4 mb = *(const f32x4*)(mp0 + 4);
      f32x4 mc = *(const f32x4*)(mp1);
      f32x4 md = *(const f32x4*)(mp1 + 4);
#pragma unroll
      for (int r = 0; r < 4; ++r) {
        sc0[r]     = s0[r]     * inv_scale + ma[r];
        sc0[r + 4] = s0[r + 4] * inv_scale + mb[r];
        sc1[r]     = s1[r]     * inv_scale + mc[r];
        sc1[r + 4] = s1[r + 4] * inv_scale + md[r];
      }
    } else {
#pragma unroll
      for (int r = 0; r < 8; ++r) {
        sc0[r] = s0[r] * inv_scale;
        sc1[r] = s1[r] * inv_scale;
      }
    }

    // ---- online softmax (per-lane over 16 keys; merge halves via xor-16) ----
    float tmax = sc0[0];
#pragma unroll
    for (int r = 1; r < 8; ++r) tmax = fmaxf(tmax, sc0[r]);
#pragma unroll
    for (int r = 0; r < 8; ++r) tmax = fmaxf(tmax, sc1[r]);
    tmax = fmaxf(tmax, __shfl_xor(tmax, 16, 32));
    float m_new = fmaxf(m_run, tmax);
    float alpha = __expf(m_run - m_new);
    float p0[8], p1[8];
    float ls = 0.f;
#pragma unroll
    for (int r = 0; r < 8; ++r) { p0[r] = __expf(sc0[r] - m_new); ls += p0[r]; }
#pragma unroll
    for (int r = 0; r < 8; ++r) { p1[r] = __expf(sc1[r] - m_new); ls += p1[r]; }
    l_run = l_run * alpha + ls;
    m_run = m_new;

    // ---- rescale O accumulator: O row r is query (8*half + r) ----
#pragma unroll
    for (int r = 0; r < 8; ++r) {
      float beta = __shfl(alpha, (half << 3) + r, 32);
#pragma unroll
      for (int c = 0; c < 8; ++c) Oacc[c][r] *= beta;
    }

    // ---- pack P (in-lane D->A layout conversion) ----
    u32x8 pw;
    pw[0] = pk_bf16(p0[0], p0[1]); pw[1] = pk_bf16(p0[2], p0[3]);
    pw[2] = pk_bf16(p0[4], p0[5]); pw[3] = pk_bf16(p0[6], p0[7]);
    pw[4] = pk_bf16(p1[0], p1[1]); pw[5] = pk_bf16(p1[2], p1[3]);
    pw[6] = pk_bf16(p1[4], p1[5]); pw[7] = pk_bf16(p1[6], p1[7]);
    v16bf Ap = __builtin_bit_cast(v16bf, pw);

    // ---- O += P x V (B operand: key-pair words 4*half..+3 and 8+4*half..+3 per lane) ----
#pragma unroll
    for (int c = 0; c < 8; ++c) {
      const unsigned int* vp = &Vb[(c * 16 + q16) * VROW + 4 * half];
      u32x4 lo = *(const u32x4*)(vp);
      u32x4 hi = *(const u32x4*)(vp + 8);
      u32x8 w  = __builtin_shufflevector(lo, hi, 0, 1, 2, 3, 4, 5, 6, 7);
      v16bf Bv = __builtin_bit_cast(v16bf, w);
      Oacc[c] = __builtin_amdgcn_wmma_f32_16x16x32_bf16(false, Ap, false, Bv,
                                                        (short)0, Oacc[c], false, false);
    }
  }

  // ---- finalize: divide by softmax denom, write rolled-back output ----
  float l_tot = l_run + __shfl_xor(l_run, 16, 32);
  float inv_l = 1.0f / l_tot;
  const int qbase = qtile * 128 + wv * 16;
#pragma unroll
  for (int r = 0; r < 8; ++r) {
    float g = __shfl(inv_l, (half << 3) + r, 32);
    int qrow_local = qbase + (half << 3) + r;
    int tok = tok_index(wy, wx, qrow_local, shift);
    float* po = Out + (bOff + tok) * HD + q16;
#pragma unroll
    for (int c = 0; c < 8; ++c) po[c * 16] = Oacc[c][r] * g;
  }
}

extern "C" void kernel_launch(void* const* d_in, const int* in_sizes, int n_in,
                              void* d_out, int out_size, void* d_ws, size_t ws_size,
                              hipStream_t stream) {
  (void)in_sizes; (void)n_in; (void)out_size; (void)d_ws; (void)ws_size;
  const float* q    = (const float*)d_in[0];
  const float* k    = (const float*)d_in[1];
  const float* v    = (const float*)d_in[2];
  const float* mask = (const float*)d_in[3];
  const int*   wsft = (const int*)d_in[5];   // with_shift scalar
  float* out = (float*)d_out;

  dim3 grid(256 * 8);   // 256 windows x 8 query tiles (128 rows each)
  dim3 block(256);      // 8 wave32 waves
  hipLaunchKernelGGL(swin_flash_attn_wmma, grid, block, 0, stream,
                     q, k, v, mask, wsft, out);
}
